// LSTM_capsule_Model_80900003987490
// MI455X (gfx1250) — compile-verified, matrix operand order checked by reading, not run on hardware
//
#include <hip/hip_runtime.h>
#include <hip/hip_bf16.h>
#include <math.h>

// ---- model dims ----
#define BATCH   32
#define SEQ     1024
#define EMBED   256
#define HIDDEN  512
#define G4H     2048      // 4*HIDDEN
#define CHG     200       // CHARGENUM
#define CAPP    64        // CAP_DIM=50 padded to 64 (pads are exact zeros)
#define CAPVALID 50

typedef __attribute__((ext_vector_type(16))) __bf16 v16bf;
typedef __attribute__((ext_vector_type(8)))  __bf16 v8bf;
typedef __attribute__((ext_vector_type(8)))  float  v8f;
typedef __attribute__((ext_vector_type(4)))  float  v4f;

__device__ __forceinline__ v8f wmma_bf16(v16bf a, v16bf b, v8f c) {
  return __builtin_amdgcn_wmma_f32_16x16x32_bf16(false, a, false, b, (short)0, c, false, false);
}

// -----------------------------------------------------------------------------
// Repack a fp32 [K, srcLd] matrix into WMMA B-fragment bf16 layout:
// dst[((kc*nTiles + nt)*32 + lane)*16 + e], with per-ISA k/lane mapping.
// cols >= srcNValid are zero (used to pad cap_W from 50 -> 64 cols).
// -----------------------------------------------------------------------------
__global__ void k_repack_frag(const float* __restrict__ src, __bf16* __restrict__ dst,
                              int kChunks, int nTiles, int srcLd, int srcNValid) {
  int idx = blockIdx.x * blockDim.x + threadIdx.x;
  int total = kChunks * nTiles * 512;
  if (idx >= total) return;
  int e    = idx & 15;
  int lane = (idx >> 4) & 31;
  int g    = idx >> 9;            // kc*nTiles + nt
  int nt   = g % nTiles;
  int kc   = g / nTiles;
  int kb   = (lane < 16) ? 0 : 8;
  int k    = kc * 32 + kb + e + (e >= 8 ? 8 : 0);
  int col  = nt * 16 + (lane & 15);
  float v  = (col < srcNValid) ? src[(size_t)k * srcLd + col] : 0.f;
  dst[idx] = (__bf16)v;
}

// -----------------------------------------------------------------------------
// Kernel 1: x_proj[s*32+b, 0:2048] = emb[enc[b,s]] @ Wx + b_lstm   (fp32 out)
// Layout [S][B][4H] so each LSTM step reads one contiguous 256 KB slab.
// Each wave: 1 M-tile x 4 N-tiles (A reused 4x), K = 256 = 8 chunks of 32.
// -----------------------------------------------------------------------------
__global__ void k_xproj(const int* __restrict__ enc, const float* __restrict__ emb,
                        const __bf16* __restrict__ WxB, const float* __restrict__ bias,
                        float* __restrict__ xproj) {
  int lane = threadIdx.x & 31;
  int gw   = blockIdx.x * 8 + (threadIdx.x >> 5);
  int mtile = gw >> 5;          // 0..2047
  int ng    = gw & 31;          // 0..31 (64-wide N group)
  int ln  = lane & 15;
  bool hi = lane >= 16;

  int rowA = mtile * 16 + ln;               // flat row = s*32 + b
  int tok  = enc[(rowA & 31) * SEQ + (rowA >> 5)];

  v8f acc[4];
#pragma unroll
  for (int j = 0; j < 4; ++j) {
    float bv = bias[ng * 64 + j * 16 + ln];
#pragma unroll
    for (int r = 0; r < 8; ++r) acc[j][r] = bv;
  }

#pragma unroll
  for (int kc = 0; kc < 8; ++kc) {
    int kb = kc * 32 + (hi ? 8 : 0);
    const float* ap = emb + (size_t)tok * EMBED + kb;
    v4f x0 = *(const v4f*)(ap + 0);
    v4f x1 = *(const v4f*)(ap + 4);
    v4f x2 = *(const v4f*)(ap + 16);
    v4f x3 = *(const v4f*)(ap + 20);
    v16bf a;
#pragma unroll
    for (int i = 0; i < 4; ++i) {
      a[i]      = (__bf16)x0[i];
      a[i + 4]  = (__bf16)x1[i];
      a[i + 8]  = (__bf16)x2[i];
      a[i + 12] = (__bf16)x3[i];
    }
#pragma unroll
    for (int j = 0; j < 4; ++j) {
      int nt = ng * 4 + j;
      v16bf bf = *(const v16bf*)(WxB + ((size_t)(kc * 128 + nt) * 32 + lane) * 16);
      acc[j] = wmma_bf16(a, bf, acc[j]);
    }
  }

#pragma unroll
  for (int j = 0; j < 4; ++j) {
#pragma unroll
    for (int r = 0; r < 8; ++r) {
      int row = mtile * 16 + r + (hi ? 8 : 0);
      xproj[(size_t)row * G4H + ng * 64 + j * 16 + ln] = acc[j][r];
    }
  }
}

// -----------------------------------------------------------------------------
// Kernel 2: persistent LSTM. 32 blocks (one per 16-wide h strip), 8 waves each.
// Wave (gate=w>>1, m=w&1) owns one 16x16 gate tile; K=512 = 16 bf16 WMMA chunks.
// h stored bf16 in Hall[t][b][h]; Hall[0] == h0 == 0. c state lives in LDS.
// Software agent-scope barrier between steps (32 co-resident blocks).
// -----------------------------------------------------------------------------
__global__ void k_lstm(const float* __restrict__ xproj, const __bf16* __restrict__ WhB,
                       __bf16* __restrict__ Hall, unsigned int* __restrict__ bar) {
  __shared__ float gLds[4][32][16];
  __shared__ float cst[32][16];

  int tid  = threadIdx.x;
  int lane = tid & 31;
  int w    = tid >> 5;
  int gate = w >> 1;
  int m    = w & 1;
  int ln   = lane & 15;
  bool hi  = lane >= 16;
  int n0   = blockIdx.x * 16;

  for (int i = tid; i < BATCH * 16; i += 256) cst[i >> 4][i & 15] = 0.f;
  __syncthreads();

  for (int t = 1; t <= SEQ; ++t) {
    // init accumulator with x_proj tile (bias already folded in)
    const float* xp = xproj + (size_t)(t - 1) * BATCH * G4H;
    int col = gate * HIDDEN + n0 + ln;
    v8f acc;
#pragma unroll
    for (int r = 0; r < 8; ++r)
      acc[r] = xp[(size_t)(m * 16 + r + (hi ? 8 : 0)) * G4H + col];

    const __bf16* hprev = Hall + (size_t)(t - 1) * BATCH * HIDDEN;
    int rowa = m * 16 + ln;
    int nt   = gate * 32 + blockIdx.x;
#pragma unroll 2
    for (int kc = 0; kc < 16; ++kc) {
      int kb = kc * 32 + (hi ? 8 : 0);
      v8bf a0 = *(const v8bf*)(hprev + (size_t)rowa * HIDDEN + kb);
      v8bf a1 = *(const v8bf*)(hprev + (size_t)rowa * HIDDEN + kb + 16);
      v16bf a;
#pragma unroll
      for (int i = 0; i < 8; ++i) { a[i] = a0[i]; a[i + 8] = a1[i]; }
      v16bf bf = *(const v16bf*)(WhB + ((size_t)(kc * 128 + nt) * 32 + lane) * 16);
      acc = wmma_bf16(a, bf, acc);
    }

#pragma unroll
    for (int r = 0; r < 8; ++r)
      gLds[gate][m * 16 + r + (hi ? 8 : 0)][ln] = acc[r];
    __syncthreads();

    // gate math + state update for this block's 32x16 strip
    __bf16* hout = Hall + (size_t)t * BATCH * HIDDEN;
    for (int i = tid; i < BATCH * 16; i += 256) {
      int row = i >> 4, c2 = i & 15;
      float iv = 1.f / (1.f + expf(-gLds[0][row][c2]));
      float fv = 1.f / (1.f + expf(-gLds[1][row][c2]));
      float gv = tanhf(gLds[2][row][c2]);
      float ov = 1.f / (1.f + expf(-gLds[3][row][c2]));
      float cc = fv * cst[row][c2] + iv * gv;
      cst[row][c2] = cc;
      hout[(size_t)row * HIDDEN + n0 + c2] = (__bf16)(ov * tanhf(cc));
    }
    __threadfence();          // make h_t visible at agent scope
    __syncthreads();
    if (tid == 0) {
      atomicAdd(bar, 1u);
      while (*(volatile unsigned int*)bar < 32u * (unsigned)t)
        __builtin_amdgcn_s_sleep(2);
    }
    __syncthreads();
    __threadfence();          // acquire side
  }
}

// -----------------------------------------------------------------------------
// Kernel 3: uji[s*32+b, 0:64] = h_t(s) @ cap_W  (cols 50..63 exactly zero)
// -----------------------------------------------------------------------------
__global__ void k_uji(const __bf16* __restrict__ Hall, const __bf16* __restrict__ capWB,
                      float* __restrict__ uji) {
  int lane = threadIdx.x & 31;
  int gw   = blockIdx.x * 8 + (threadIdx.x >> 5);
  int mtile = gw >> 2;          // 0..2047
  int nt    = gw & 3;           // 0..3
  int ln  = lane & 15;
  bool hi = lane >= 16;

  v8f acc;
#pragma unroll
  for (int r = 0; r < 8; ++r) acc[r] = 0.f;

  int row = mtile * 16 + ln;                        // s*32 + b
  const __bf16* ap = Hall + (size_t)(row + BATCH) * HIDDEN;  // en_outputs row = Hall[s+1][b]
#pragma unroll 2
  for (int kc = 0; kc < 16; ++kc) {
    int kb = kc * 32 + (hi ? 8 : 0);
    v8bf a0 = *(const v8bf*)(ap + kb);
    v8bf a1 = *(const v8bf*)(ap + kb + 16);
    v16bf a;
#pragma unroll
    for (int i = 0; i < 8; ++i) { a[i] = a0[i]; a[i + 8] = a1[i]; }
    v16bf bf = *(const v16bf*)(capWB + ((size_t)(kc * 4 + nt) * 32 + lane) * 16);
    acc = wmma_bf16(a, bf, acc);
  }
#pragma unroll
  for (int r = 0; r < 8; ++r)
    uji[(size_t)(mtile * 16 + r + (hi ? 8 : 0)) * CAPP + nt * 16 + ln] = acc[r];
}

// -----------------------------------------------------------------------------
// Routing: Z[b,n] = sum_s exp(B[b,n,s]) * mask[b,s]      (routeB layout [b][n][s])
// -----------------------------------------------------------------------------
__global__ void k_Z(const float* __restrict__ routeB, const int* __restrict__ enc,
                    float* __restrict__ Z) {
  __shared__ float red[256];
  int bn = blockIdx.x;              // b*200 + n
  int b  = bn / CHG;
  float acc = 0.f;
  for (int s = threadIdx.x; s < SEQ; s += 256) {
    float mk = (enc[b * SEQ + s] > 0) ? 1.f : 0.f;
    acc += expf(routeB[(size_t)bn * SEQ + s]) * mk;
  }
  red[threadIdx.x] = acc;
  __syncthreads();
  for (int off = 128; off > 0; off >>= 1) {
    if (threadIdx.x < off) red[threadIdx.x] += red[threadIdx.x + off];
    __syncthreads();
  }
  if (threadIdx.x == 0) Z[bn] = red[0];
}

// -----------------------------------------------------------------------------
// Routing: s_cap[b,n,:] = sum_s c[b,s,n] * uji[b,s,:], then squash -> v[b,n,:]
// 64 threads per block (p dimension), norm over 64 (pads are zero).
// -----------------------------------------------------------------------------
__global__ void k_SV(const float* __restrict__ routeB, const int* __restrict__ enc,
                     const float* __restrict__ Z, const float* __restrict__ uji,
                     float* __restrict__ v) {
  __shared__ float red[64];
  int bn = blockIdx.x;
  int b  = bn / CHG;
  int p  = threadIdx.x;
  float zinv = 1.f / Z[bn];
  float acc = 0.f;
  for (int s = 0; s < SEQ; ++s) {
    float bb = routeB[(size_t)bn * SEQ + s];             // lane-uniform broadcast
    float mk = (enc[b * SEQ + s] > 0) ? 1.f : 0.f;
    float cv = expf(bb) * mk * zinv;
    acc += cv * uji[((size_t)s * BATCH + b) * CAPP + p];
  }
  red[p] = acc * acc;
  __syncthreads();
  for (int off = 32; off > 0; off >>= 1) {
    if (p < off) red[p] += red[p + off];
    __syncthreads();
  }
  float n2 = red[0];
  float nr = sqrtf(n2);
  v[(size_t)bn * CAPP + p] = (nr / (1.f + n2)) * acc;
}

// -----------------------------------------------------------------------------
// Routing: b_new[b,n,s] = sum_p v[b,n,p] * uji[b,s,p]
// -----------------------------------------------------------------------------
__global__ void k_D(const float* __restrict__ v, const float* __restrict__ uji,
                    float* __restrict__ routeB) {
  __shared__ float vsh[CAPP];
  int bn = blockIdx.x;
  int b  = bn / CHG;
  if (threadIdx.x < CAPP) vsh[threadIdx.x] = v[(size_t)bn * CAPP + threadIdx.x];
  __syncthreads();
  for (int s = threadIdx.x; s < SEQ; s += 256) {
    const float* up = uji + ((size_t)s * BATCH + b) * CAPP;
    float d = 0.f;
#pragma unroll
    for (int p = 0; p < CAPP; ++p) d += vsh[p] * up[p];
    routeB[(size_t)bn * SEQ + s] = d;
  }
}

// -----------------------------------------------------------------------------
// Margin loss -> scalar mean
// -----------------------------------------------------------------------------
__global__ void k_loss(const float* __restrict__ v, const int* __restrict__ labels,
                       float* __restrict__ out) {
  __shared__ float red[256];
  float acc = 0.f;
  for (int idx = threadIdx.x; idx < BATCH * CHG; idx += 256) {
    int b = idx / CHG, n = idx % CHG;
    const float* vp = v + (size_t)idx * CAPP;
    float n2 = 0.f;
#pragma unroll
    for (int p = 0; p < CAPP; ++p) n2 += vp[p] * vp[p];
    float nr  = sqrtf(n2);
    float pos = fmaxf(0.9f - nr, 0.f); pos *= pos;
    float neg = fmaxf(nr - 0.1f, 0.f); neg *= neg;
    float oh  = (labels[b] == n) ? 1.f : 0.f;
    acc += oh * pos + 0.5f * (1.f - oh) * neg;
  }
  red[threadIdx.x] = acc;
  __syncthreads();
  for (int off = 128; off > 0; off >>= 1) {
    if (threadIdx.x < off) red[threadIdx.x] += red[threadIdx.x + off];
    __syncthreads();
  }
  if (threadIdx.x == 0) out[0] = red[0] / (float)BATCH;
}

// -----------------------------------------------------------------------------
extern "C" void kernel_launch(void* const* d_in, const int* in_sizes, int n_in,
                              void* d_out, int out_size, void* d_ws, size_t ws_size,
                              hipStream_t stream) {
  const int*   enc    = (const int*)d_in[0];
  const int*   labels = (const int*)d_in[1];
  const float* emb    = (const float*)d_in[2];
  const float* Wx     = (const float*)d_in[3];
  const float* Wh     = (const float*)d_in[4];
  const float* b_lstm = (const float*)d_in[5];
  const float* cap_W  = (const float*)d_in[6];

  char* ws = (char*)d_ws;
  auto alloc = [&](size_t bytes) -> char* {
    char* p = ws;
    ws += (bytes + 255) & ~(size_t)255;
    return p;
  };
  float*        xproj  = (float*)alloc((size_t)SEQ * BATCH * G4H * 4);            // 256 MB
  __bf16*       Hall   = (__bf16*)alloc((size_t)(SEQ + 1) * BATCH * HIDDEN * 2);  // 33.6 MB
  __bf16*       WxB    = (__bf16*)alloc((size_t)EMBED * G4H * 2);                 // 1 MB
  __bf16*       WhB    = (__bf16*)alloc((size_t)HIDDEN * G4H * 2);                // 2 MB
  __bf16*       capWB  = (__bf16*)alloc((size_t)HIDDEN * CAPP * 2);               // 64 KB
  float*        uji    = (float*)alloc((size_t)SEQ * BATCH * CAPP * 4);           // 8 MB
  float*        routeB = (float*)alloc((size_t)BATCH * CHG * SEQ * 4);            // 26 MB
  float*        Zbuf   = (float*)alloc((size_t)BATCH * CHG * 4);
  float*        vbuf   = (float*)alloc((size_t)BATCH * CHG * CAPP * 4);
  unsigned int* bar    = (unsigned int*)alloc(256);

  // zero: routing logits, h0, barrier counter
  hipMemsetAsync(routeB, 0, (size_t)BATCH * CHG * SEQ * 4, stream);
  hipMemsetAsync(Hall, 0, (size_t)BATCH * HIDDEN * 2, stream);
  hipMemsetAsync(bar, 0, 256, stream);

  // repack weights into WMMA bf16 fragment layout
  k_repack_frag<<<(8 * 128 * 512 + 255) / 256, 256, 0, stream>>>(Wx, WxB, 8, 128, G4H, G4H);
  k_repack_frag<<<(16 * 128 * 512 + 255) / 256, 256, 0, stream>>>(Wh, WhB, 16, 128, G4H, G4H);
  k_repack_frag<<<(16 * 4 * 512 + 255) / 256, 256, 0, stream>>>(cap_W, capWB, 16, 4, CAPVALID, CAPVALID);

  // embed gather + input projection GEMM (bf16 WMMA, fp32 out)
  k_xproj<<<8192, 256, 0, stream>>>(enc, emb, WxB, b_lstm, xproj);

  // persistent LSTM over 1024 steps
  k_lstm<<<32, 256, 0, stream>>>(xproj, WhB, Hall, bar);

  // capsule projection GEMM
  k_uji<<<1024, 256, 0, stream>>>(Hall, capWB, uji);

  // 3 routing iterations (skip final b-update, unused)
  for (int it = 0; it < 3; ++it) {
    k_Z<<<BATCH * CHG, 256, 0, stream>>>(routeB, enc, Zbuf);
    k_SV<<<BATCH * CHG, 64, 0, stream>>>(routeB, enc, Zbuf, uji, vbuf);
    if (it < 2) k_D<<<BATCH * CHG, 256, 0, stream>>>(vbuf, uji, routeB);
  }

  k_loss<<<1, 256, 0, stream>>>(vbuf, labels, (float*)d_out);

  (void)in_sizes; (void)n_in; (void)out_size; (void)ws_size;
}
